// PCQSqueezeExcitation_55439437857267
// MI455X (gfx1250) — matrix operand order repaired and checked by reading, not compiled, not run on hardware
//
#include <hip/hip_runtime.h>
#include <hip/hip_bf16.h>
#include <math.h>

// ---------------------------------------------------------------------------
// PCQ Squeeze-Excitation for MI455X (gfx1250, wave32)
//   K1: per-(b,c) plane sum/min/max reduction (one pass over x)
//   K2: SE MLP via v_wmma_f32_16x16x32_f16 + cluster EMA quant params
//   K3: streaming fake-quantize pass (one read of x, one write of out)
// ---------------------------------------------------------------------------

#define B_  64
#define C_  256
#define CS_ 64
#define HW_ 4096        // 64*64
#define NC_ 8
#define QMAX_ 255.0f
#define SMOOTH_ 0.995f

typedef __attribute__((ext_vector_type(16))) _Float16 v16h;
typedef __attribute__((ext_vector_type(8)))  float    v8f;
typedef __attribute__((ext_vector_type(4)))  float    f32x4;

// Workspace layout (floats)
#define WS_POOLED 0
#define WS_XMIN   (B_*C_)
#define WS_XMAX   (2*B_*C_)
#define WS_SCALE  (3*B_*C_)
#define WS_QS     (4*B_*C_)
#define WS_QZ     (4*B_*C_ + NC_)

// A-matrix (16-bit, 16x32) K index for element e (0..15), lane group g (0..1)
// ISA 7.12.2: VGPR v<4: K=2v+lo+8g ; v>=4: K=16+2(v-4)+lo+8g
__device__ __forceinline__ int a_k_of(int e, int g) {
    int v = e >> 1, lo = e & 1;
    return (v < 4) ? (2 * v + lo + 8 * g) : (16 + 2 * (v - 4) + lo + 8 * g);
}
// B-matrix (16-bit, 32x16): same K order as A (e<8: K=e+8g ; e>=8: K=16+(e-8)+8g)
__device__ __forceinline__ int b_k_of(int e, int g) {
    return a_k_of(e, g);
}

// ---------------------------------------------------------------------------
// Kernel 1: per-plane (b,c) sum / min / max of x.  16384 blocks x 256 threads.
// ---------------------------------------------------------------------------
__global__ void __launch_bounds__(256)
k_pool_minmax(const float* __restrict__ x, float* __restrict__ ws) {
    const int plane = blockIdx.x;                       // b*C + c
    const f32x4* px = (const f32x4*)(x + (size_t)plane * HW_);

    float s = 0.0f, mn = __builtin_inff(), mx = -__builtin_inff();
#pragma unroll
    for (int i = 0; i < 4; ++i) {
        f32x4 v = __builtin_nontemporal_load(&px[i * 256 + threadIdx.x]);
        s += (v.x + v.y) + (v.z + v.w);
        mn = fminf(mn, fminf(fminf(v.x, v.y), fminf(v.z, v.w)));
        mx = fmaxf(mx, fmaxf(fmaxf(v.x, v.y), fmaxf(v.z, v.w)));
    }
    // wave32 reduction
#pragma unroll
    for (int off = 16; off >= 1; off >>= 1) {
        s += __shfl_xor(s, off, 32);
        mn = fminf(mn, __shfl_xor(mn, off, 32));
        mx = fmaxf(mx, __shfl_xor(mx, off, 32));
    }
    __shared__ float ls[8], lmn[8], lmx[8];
    const int wv = threadIdx.x >> 5;
    if ((threadIdx.x & 31) == 0) { ls[wv] = s; lmn[wv] = mn; lmx[wv] = mx; }
    __syncthreads();
    if (threadIdx.x == 0) {
        float S = 0.0f, MN = __builtin_inff(), MX = -__builtin_inff();
#pragma unroll
        for (int i = 0; i < 8; ++i) {
            S += ls[i]; MN = fminf(MN, lmn[i]); MX = fmaxf(MX, lmx[i]);
        }
        ws[WS_POOLED + plane] = S * (1.0f / (float)HW_);
        ws[WS_XMIN + plane] = MN;
        ws[WS_XMAX + plane] = MX;
    }
}

// ---------------------------------------------------------------------------
// Kernel 2: SE MLP with WMMA + cluster EMA quant params.
// 1 block, 512 threads (16 waves).
// ---------------------------------------------------------------------------
__global__ void __launch_bounds__(512)
k_se_wmma(const float* __restrict__ w1, const float* __restrict__ b1,
          const float* __restrict__ w2, const float* __restrict__ b2,
          const float* __restrict__ act_range,
          const int* __restrict__ sample_cluster,
          float* __restrict__ ws) {
    __shared__ float hbuf[B_ * CS_];          // 16 KB: h = relu(pooled@w1^T+b1)
    __shared__ float pmin[8 * B_], pmax[8 * B_];
    __shared__ float sminL[B_], smaxL[B_];

    const float* pooled = ws + WS_POOLED;
    const int lane = threadIdx.x & 31;
    const int w    = threadIdx.x >> 5;        // wave id 0..15
    const int ml   = lane & 15;               // M (or N) within tile
    const int g    = lane >> 4;               // lane group

    // ---- Stage 1: h[64,64] = relu(pooled[64,256] @ w1[64,256]^T + b1) ----
    {
        const int mi = w >> 2;                // b-tile 0..3
        const int ni = w & 3;                 // s-tile 0..3
        v8f acc = {};
#pragma unroll
        for (int k0 = 0; k0 < C_; k0 += 32) {
            v16h a, b;
#pragma unroll
            for (int e = 0; e < 16; ++e) {
                a[e] = (_Float16)pooled[(mi * 16 + ml) * C_ + k0 + a_k_of(e, g)];
                b[e] = (_Float16)w1[(ni * 16 + ml) * C_ + k0 + b_k_of(e, g)];
            }
            acc = __builtin_amdgcn_wmma_f32_16x16x32_f16(
                false, a, false, b, (short)0, acc, false, false);
        }
        const int n = ni * 16 + ml;           // s index
        const float bias1 = b1[n];
#pragma unroll
        for (int r = 0; r < 8; ++r) {
            const int m = mi * 16 + r + 8 * g; // b index
            hbuf[m * CS_ + n] = fmaxf(acc[r] + bias1, 0.0f);
        }
    }
    __syncthreads();

    // ---- Stage 2: v[64,256] = h @ w2[256,64]^T + b2 ; scale=hardsigmoid ----
    {
        const int mi2 = w >> 2;               // b-tile 0..3
#pragma unroll
        for (int j = 0; j < 4; ++j) {
            const int ni2 = (w & 3) * 4 + j;  // c-tile 0..15
            v8f acc = {};
#pragma unroll
            for (int k0 = 0; k0 < CS_; k0 += 32) {
                v16h a, b;
#pragma unroll
                for (int e = 0; e < 16; ++e) {
                    a[e] = (_Float16)hbuf[(mi2 * 16 + ml) * CS_ + k0 + a_k_of(e, g)];
                    b[e] = (_Float16)w2[(ni2 * 16 + ml) * CS_ + k0 + b_k_of(e, g)];
                }
                acc = __builtin_amdgcn_wmma_f32_16x16x32_f16(
                    false, a, false, b, (short)0, acc, false, false);
            }
            const int c = ni2 * 16 + ml;
            const float bias2 = b2[c];
#pragma unroll
            for (int r = 0; r < 8; ++r) {
                const int bidx = mi2 * 16 + r + 8 * g;
                const float v = acc[r] + bias2;
                const float sc = fminf(fmaxf(v * (1.0f / 6.0f) + 0.5f, 0.0f), 1.0f);
                ws[WS_SCALE + bidx * C_ + c] = sc;
            }
        }
    }
    __threadfence();
    __syncthreads();

    // ---- Stage 3: per-sample min/max of out via scale*xmin / scale*xmax ----
    // (scale >= 0, so min_hw(scale*x) = scale*xmin, max_hw = scale*xmax)
    {
        const int b  = threadIdx.x & 63;
        const int gr = threadIdx.x >> 6;      // 0..7
        float mn = __builtin_inff(), mx = -__builtin_inff();
        for (int c = gr * 32; c < gr * 32 + 32; ++c) {
            const float sc = ws[WS_SCALE + b * C_ + c];
            mn = fminf(mn, sc * ws[WS_XMIN + b * C_ + c]);
            mx = fmaxf(mx, sc * ws[WS_XMAX + b * C_ + c]);
        }
        pmin[gr * B_ + b] = mn;
        pmax[gr * B_ + b] = mx;
    }
    __syncthreads();
    if (threadIdx.x < B_) {
        const int b = threadIdx.x;
        float mn = __builtin_inff(), mx = -__builtin_inff();
#pragma unroll
        for (int gr = 0; gr < 8; ++gr) {
            mn = fminf(mn, pmin[gr * B_ + b]);
            mx = fmaxf(mx, pmax[gr * B_ + b]);
        }
        sminL[b] = mn;
        smaxL[b] = mx;
    }
    __syncthreads();

    // ---- Stage 4: segment min/max per cluster + EMA + (s,z) ----
    if (threadIdx.x < NC_) {
        const int nc = threadIdx.x;
        float cmin = __builtin_inff(), cmax = -__builtin_inff();
        for (int b = 0; b < B_; ++b) {
            if (sample_cluster[b] == nc) {
                cmin = fminf(cmin, sminL[b]);
                cmax = fmaxf(cmax, smaxL[b]);
            }
        }
        const float nmin = act_range[nc * 2 + 0] * SMOOTH_ + cmin * (1.0f - SMOOTH_);
        const float nmax = act_range[nc * 2 + 1] * SMOOTH_ + cmax * (1.0f - SMOOTH_);
        const float s = (nmax - nmin) * (1.0f / QMAX_);
        const float z = -rintf(nmin / s);
        ws[WS_QS + nc] = s;
        ws[WS_QZ + nc] = z;
    }
}

// ---------------------------------------------------------------------------
// Kernel 3: streaming fake-quantize.  out = (clamp(rint(o/s+z),0,255)-z)*s
// 65536 blocks x 256 threads, one float4 each.
// ---------------------------------------------------------------------------
__global__ void __launch_bounds__(256)
k_quant(const float* __restrict__ x, const float* __restrict__ ws,
        const int* __restrict__ sample_cluster, float* __restrict__ out) {
    const int idx4 = blockIdx.x * 256 + threadIdx.x;   // < 2^24
    const int base = idx4 << 2;                        // element index, < 2^26
    const int b = base >> 20;                          // / (C*H*W) = / 2^20
    const int c = (base >> 12) & (C_ - 1);             // / (H*W) mod C

    const float sc = ws[WS_SCALE + b * C_ + c];
    const int nc = sample_cluster[b];
    const float s = ws[WS_QS + nc];
    const float z = ws[WS_QZ + nc];

    f32x4 xv = __builtin_nontemporal_load(&((const f32x4*)x)[idx4]);
    f32x4 r;
#pragma unroll
    for (int i = 0; i < 4; ++i) {
        const float o = sc * xv[i];
        const float q = fminf(fmaxf(rintf(o / s + z), 0.0f), QMAX_);
        r[i] = (q - z) * s;
    }
    __builtin_nontemporal_store(r, &((f32x4*)out)[idx4]);
}

// ---------------------------------------------------------------------------
extern "C" void kernel_launch(void* const* d_in, const int* in_sizes, int n_in,
                              void* d_out, int out_size, void* d_ws, size_t ws_size,
                              hipStream_t stream) {
    const float* x   = (const float*)d_in[0];
    const float* w1  = (const float*)d_in[1];
    const float* b1  = (const float*)d_in[2];
    const float* w2  = (const float*)d_in[3];
    const float* b2  = (const float*)d_in[4];
    const float* ar  = (const float*)d_in[5];
    const int*   sc  = (const int*)d_in[6];
    float* out = (float*)d_out;
    float* ws  = (float*)d_ws;

    // K1: 16384 planes (one block each)
    k_pool_minmax<<<B_ * C_, 256, 0, stream>>>(x, ws);
    // K2: single workgroup, 16 waves, WMMA SE MLP + quant params
    k_se_wmma<<<1, 512, 0, stream>>>(w1, b1, w2, b2, ar, sc, ws);
    // K3: streaming quantize, 64M elements / 4 per thread
    k_quant<<<(B_ * C_ * HW_) / (4 * 256), 256, 0, stream>>>(x, ws, sc, out);
}
